// ODEDecoder_6296422056489
// MI455X (gfx1250) — matrix operand order repaired and checked by reading, not compiled, not run on hardware
//
#include <hip/hip_runtime.h>
#include <hip/hip_bf16.h>
#include <math.h>

// ---------------------------------------------------------------------------
// ODE decoder for MI455X (gfx1250, wave32, WMMA).
// All GEMMs run as v_wmma_f32_16x16x32_bf16 (f32 accumulate).
// Weights are pre-packed into WMMA B-fragment layout; activations stay in
// LDS / registers across the whole 48-stage dopri5 integration.
// ---------------------------------------------------------------------------

typedef __attribute__((ext_vector_type(16))) __bf16 v16bf;
typedef __attribute__((ext_vector_type(8)))  float  v8f;

#define B_EDGES 192
#define L_NODES 64
#define ZD 128
#define HD 256
#define XD 172
#define KPAD_D 320            // decode K padded 300 -> 320
#define NROWS (B_EDGES * L_NODES)   // 12288
#define NSTEPS 8
#define DTC (1.0f / 8.0f)

__device__ __constant__ float d_AC[5][5] = {
  {0.2f, 0.f, 0.f, 0.f, 0.f},
  {3.f/40.f, 9.f/40.f, 0.f, 0.f, 0.f},
  {44.f/45.f, -56.f/15.f, 32.f/9.f, 0.f, 0.f},
  {19372.f/6561.f, -25360.f/2187.f, 64448.f/6561.f, -212.f/729.f, 0.f},
  {9017.f/3168.f, -355.f/33.f, 46732.f/5247.f, 49.f/176.f, -5103.f/18656.f}
};
__device__ __constant__ float d_CC[6] = {0.f, 0.2f, 0.3f, 0.8f, 8.f/9.f, 1.f};

// bf16 <-> f32 via bit ops (round-to-nearest-even).
__device__ __forceinline__ __bf16 f2bf(float f) {
  union { float f; unsigned u; } a; a.f = f;
  unsigned r = a.u + 0x7FFFu + ((a.u >> 16) & 1u);
  unsigned short h = (unsigned short)(r >> 16);
  __bf16 b; __builtin_memcpy(&b, &h, 2); return b;
}
__device__ __forceinline__ float bf2f(__bf16 b) {
  unsigned short h; __builtin_memcpy(&h, &b, 2);
  union { unsigned u; float f; } a; a.u = ((unsigned)h) << 16; return a.f;
}

// Branch-free tanh: 1 - 2/(exp2(2x*log2e)+1).  v_exp_f32 + v_rcp_f32, no EXEC
// divergence; saturates to +/-1 for large |x| (exp2 -> inf/0).
__device__ __forceinline__ float fast_tanh(float x) {
  float e = __builtin_amdgcn_exp2f(x * 2.88539008177793f); // 2*log2(e)*x
  return 1.f - 2.f * __builtin_amdgcn_rcpf(e + 1.f);
}

union FragU { v16bf v; uint4 q[2]; };

// A-fragment (16x32 bf16, M x K) from a row-major [16][stride] bf16 buffer.
// Lane l: row = l%16; K-chunk {k0..k0+15}, k0 = kt*32 + (l>>4)*16.
__device__ __forceinline__ v16bf load_a(const __bf16* buf, int stride, int kt, int lane) {
  const __bf16* p = buf + (lane & 15) * stride + kt * 32 + (lane >> 4) * 16;
  FragU f;
  f.q[0] = *(const uint4*)p;
  f.q[1] = *(const uint4*)(p + 8);
  return f.v;
}

// B-fragment (32x16 bf16, K x N) from pre-packed storage: 16 contiguous bf16/lane.
__device__ __forceinline__ v16bf load_b(const __bf16* p) {
  FragU f;
  f.q[0] = *(const uint4*)p;
  f.q[1] = *(const uint4*)(p + 8);
  return f.v;
}

#define WMMA_BF16(a, b, c) \
  __builtin_amdgcn_wmma_f32_16x16x32_bf16(false, (a), false, (b), (short)0, (c), false, false)

// ---------------------------------------------------------------------------
// Pack a row-major f32 weight [Korig][N] into WMMA B-fragment bf16 layout.
// Fragment f = kt*(N/16)+nt holds 512 bf16: lane l -> col nt*16+(l&15),
// rows kt*32+(l>>4)*16 .. +15, contiguous per lane.
// ---------------------------------------------------------------------------
__global__ void pack_b_kernel(const float* __restrict__ src, __bf16* __restrict__ dst,
                              int Korig, int Kpad, int N) {
  int gid = blockIdx.x * blockDim.x + threadIdx.x;
  int frags = (Kpad >> 5) * (N >> 4);
  if (gid >= frags * 32) return;
  int f = gid >> 5, lane = gid & 31;
  int ntiles = N >> 4;
  int kt = f / ntiles, nt = f % ntiles;
  int n = nt * 16 + (lane & 15);
  int k0 = kt * 32 + (lane >> 4) * 16;
  #pragma unroll
  for (int j = 0; j < 16; ++j) {
    int kk = k0 + j;
    float v = (kk < Korig) ? src[kk * N + n] : 0.f;
    dst[f * 512 + lane * 16 + j] = f2bf(v);
  }
}

// Fill feat[:, 0:172] = x[b] (broadcast over L) and feat[:, 300:320] = 0.
__global__ void featx_kernel(const float* __restrict__ x, __bf16* __restrict__ gFeat) {
  int idx = blockIdx.x * blockDim.x + threadIdx.x;
  if (idx >= NROWS * 192) return;
  int r = idx / 192, c = idx % 192;
  if (c < XD) gFeat[r * KPAD_D + c] = f2bf(x[(r >> 6) * XD + c]);
  else        gFeat[r * KPAD_D + 300 + (c - XD)] = f2bf(0.f);
}

// ---------------------------------------------------------------------------
// ODE kernel: 192 blocks x 128 threads (4 waves, 16 rows each).
// LDS: packed W2 (128KB) + per-wave yi/h2 (8KB), h1 (8KB), k1..k5 (20KB) = 272KB.
// ---------------------------------------------------------------------------
__global__ __launch_bounds__(128)
void ode_kernel(const float* __restrict__ z, const float* __restrict__ ts,
                const float* __restrict__ w_t, const float* __restrict__ b_t,
                const float* __restrict__ b1, const float* __restrict__ b2,
                const float* __restrict__ b3,
                const __bf16* __restrict__ gW1p, const __bf16* __restrict__ gW2p,
                const __bf16* __restrict__ gW3p, __bf16* __restrict__ gFeat) {
  __shared__ __bf16 sW2[HD * HD];        // packed W2, B-fragment order
  __shared__ __bf16 sA[4][16 * HD];      // ybuf (stride ZD) then h2 (stride HD)
  __shared__ __bf16 sH1[4][16 * HD];     // h1 (stride HD)
  __shared__ __bf16 sK[4][5][16 * ZD];   // k1..k5 per wave, bf16

  const int tid = threadIdx.x;
  const int wave = tid >> 5, lane = tid & 31;
  const int col = lane & 15, half = lane >> 4;

  // Warm near caches with the streamed W1/W3 fragment buffers (64KB each):
  // emits gfx1250 global_prefetch_b8, one line per 128B.
  for (int i = tid * 64; i < 32768; i += 128 * 64) {
    __builtin_prefetch(gW1p + i, 0, 3);
    __builtin_prefetch(gW3p + i, 0, 3);
  }

  { // cooperative LDS fill of W2 (128KB)
    const uint4* src = (const uint4*)gW2p;
    uint4* dst = (uint4*)sW2;
    for (int i = tid; i < (HD * HD * 2) / 16; i += 128) dst[i] = src[i];
  }
  __syncthreads();

  const int row_base = blockIdx.x * 64 + wave * 16;

  float wt[8], bt[8], rr[8];
  #pragma unroll
  for (int t = 0; t < 8; ++t) { wt[t] = w_t[t * 16 + col]; bt[t] = b_t[t * 16 + col]; }
  #pragma unroll
  for (int v = 0; v < 8; ++v) rr[v] = ts[row_base + half * 8 + v]; // RESIZE = 1

  // y state in registers, WMMA C/D layout: y[ztile][v] <-> (row v+8*half, col 16t+col)
  float y[8][8];
  #pragma unroll
  for (int t = 0; t < 8; ++t)
    #pragma unroll
    for (int v = 0; v < 8; ++v) {
      int r = row_base + half * 8 + v;
      y[t][v] = z[(r & 63) * ZD + t * 16 + col];
    }

  __bf16* myA = sA[wave];
  __bf16* myH1 = sH1[wave];

  #pragma unroll 1
  for (int step = 0; step < NSTEPS; ++step) {
    float s0 = step * DTC;
    #pragma unroll 1
    for (int stage = 0; stage < 6; ++stage) {
      float tf = s0 + d_CC[stage] * DTC;
      // yi = y + DT*sum(a*k) + time-encoding, written row-major bf16 (stride ZD)
      #pragma unroll
      for (int t = 0; t < 8; ++t) {
        #pragma unroll
        for (int v = 0; v < 8; ++v) {
          int m = half * 8 + v;
          int off = m * ZD + t * 16 + col;
          float yi = y[t][v];
          for (int j = 0; j < stage; ++j)
            yi += DTC * d_AC[stage - 1][j] * bf2f(sK[wave][j][off]);
          float te = __cosf(tf * rr[v] * wt[t] + bt[t]);  // native v_cos_f32
          myA[off] = f2bf(yi + te);
        }
      }
      // GEMM1: h1 = tanh(yi @ W1 + b1)   [16xZD]x[ZDxHD]
      #pragma unroll 1
      for (int nt = 0; nt < 16; ++nt) {
        v8f acc = {};
        #pragma unroll
        for (int kt = 0; kt < 4; ++kt) {
          v16bf a = load_a(myA, ZD, kt, lane);
          v16bf b = load_b(gW1p + (kt * 16 + nt) * 512 + lane * 16);
          acc = WMMA_BF16(a, b, acc);
        }
        float bv = b1[nt * 16 + col];
        #pragma unroll
        for (int v = 0; v < 8; ++v)
          myH1[(half * 8 + v) * HD + nt * 16 + col] = f2bf(fast_tanh(acc[v] + bv));
      }
      // GEMM2: h2 = tanh(h1 @ W2 + b2)   [16xHD]x[HDxHD], W2 from LDS
      #pragma unroll 1
      for (int nt = 0; nt < 16; ++nt) {
        v8f acc = {};
        #pragma unroll
        for (int kt = 0; kt < 8; ++kt) {
          v16bf a = load_a(myH1, HD, kt, lane);
          v16bf b = load_b(sW2 + (kt * 16 + nt) * 512 + lane * 16);
          acc = WMMA_BF16(a, b, acc);
        }
        float bv = b2[nt * 16 + col];
        #pragma unroll
        for (int v = 0; v < 8; ++v)
          myA[(half * 8 + v) * HD + nt * 16 + col] = f2bf(fast_tanh(acc[v] + bv));
      }
      // GEMM3: k = (h2 @ W3 + b3) * ratio  [16xHD]x[HDxZD]
      #pragma unroll
      for (int nt = 0; nt < 8; ++nt) {
        v8f acc = {};
        #pragma unroll
        for (int kt = 0; kt < 8; ++kt) {
          v16bf a = load_a(myA, HD, kt, lane);
          v16bf b = load_b(gW3p + (kt * 8 + nt) * 512 + lane * 16);
          acc = WMMA_BF16(a, b, acc);
        }
        float bv = b3[nt * 16 + col];
        #pragma unroll
        for (int v = 0; v < 8; ++v) {
          int m = half * 8 + v;
          int off = m * ZD + nt * 16 + col;
          float kv = (acc[v] + bv) * rr[v];
          if (stage < 5) {
            sK[wave][stage][off] = f2bf(kv);
          } else {
            // y += DT*(35/384 k1 + 500/1113 k3 + 125/192 k4 - 2187/6784 k5 + 11/84 k6)
            y[nt][v] += DTC * (0.09114583333f * bf2f(sK[wave][0][off])
                             + 0.44923629829f * bf2f(sK[wave][2][off])
                             + 0.65104166667f * bf2f(sK[wave][3][off])
                             - 0.32237813885f * bf2f(sK[wave][4][off])
                             + 0.13095238095f * kv);
          }
        }
      }
    }
  }
  // z_final -> feat columns [172, 300)
  #pragma unroll
  for (int t = 0; t < 8; ++t)
    #pragma unroll
    for (int v = 0; v < 8; ++v) {
      int r = row_base + half * 8 + v;
      gFeat[r * KPAD_D + XD + t * 16 + col] = f2bf(y[t][v]);
    }
}

// ---------------------------------------------------------------------------
// Decode: h = feat @ Wd + bd   (12288 x 320pad x 256), h stored bf16.
// ---------------------------------------------------------------------------
__global__ __launch_bounds__(128)
void decode_kernel(const __bf16* __restrict__ gFeat, const __bf16* __restrict__ gWdp,
                   const float* __restrict__ bd, __bf16* __restrict__ gH) {
  __shared__ __bf16 sWd[(KPAD_D / 32) * 16 * 512]; // 160KB packed Wd
  const int tid = threadIdx.x;
  { const uint4* src = (const uint4*)gWdp;
    uint4* dst = (uint4*)sWd;
    for (int i = tid; i < (KPAD_D * HD * 2) / 16; i += 128) dst[i] = src[i]; }
  __syncthreads();

  const int wave = tid >> 5, lane = tid & 31;
  const int col = lane & 15, half = lane >> 4;
  const int row_base = blockIdx.x * 64 + wave * 16;
  const __bf16* aBase = gFeat + row_base * KPAD_D;

  #pragma unroll 1
  for (int nt = 0; nt < 16; ++nt) {
    v8f acc = {};
    #pragma unroll
    for (int kt = 0; kt < KPAD_D / 32; ++kt) {
      v16bf a = load_a(aBase, KPAD_D, kt, lane);
      v16bf b = load_b(sWd + (kt * 16 + nt) * 512 + lane * 16);
      acc = WMMA_BF16(a, b, acc);
    }
    float bv = bd[nt * 16 + col];
    #pragma unroll
    for (int v = 0; v < 8; ++v)
      gH[(row_base + half * 8 + v) * HD + nt * 16 + col] = f2bf(acc[v] + bv);
  }
}

// ---------------------------------------------------------------------------
// Edge predictor: one block per output group g (64 edge rows).
// pos = mean_l relu(h_src@Ws+bs + h_pos@Wdst+bdst) @ Wo + bo, likewise neg.
// ---------------------------------------------------------------------------
__global__ __launch_bounds__(128)
void edge_kernel(const __bf16* __restrict__ gH, const __bf16* __restrict__ gWsp,
                 const __bf16* __restrict__ gWdstp,
                 const float* __restrict__ bs, const float* __restrict__ bdst,
                 const float* __restrict__ Wo, const float* __restrict__ bo,
                 float* __restrict__ out) {
  __shared__ __bf16 sWs[HD * HD];    // 128KB
  __shared__ __bf16 sWd2[HD * HD];   // 128KB
  __shared__ float sRed[2][64];
  const int tid = threadIdx.x;
  { const uint4* s1 = (const uint4*)gWsp;   uint4* d1 = (uint4*)sWs;
    const uint4* s2 = (const uint4*)gWdstp; uint4* d2 = (uint4*)sWd2;
    for (int i = tid; i < (HD * HD * 2) / 16; i += 128) { d1[i] = s1[i]; d2[i] = s2[i]; } }
  __syncthreads();

  const int wave = tid >> 5, lane = tid & 31;
  const int col = lane & 15, half = lane >> 4;
  const int e_base = blockIdx.x * 64 + wave * 16;
  const __bf16* aS = gH + e_base * HD;
  const __bf16* aP = gH + (4096 + e_base) * HD;
  const __bf16* aN = gH + (8192 + e_base) * HD;

  float pacc[8] = {0}, nacc[8] = {0};
  #pragma unroll 1
  for (int nt = 0; nt < 16; ++nt) {
    v8f as = {}, ap = {}, an = {};
    #pragma unroll
    for (int kt = 0; kt < 8; ++kt) {
      v16bf bS = load_b(sWs  + (kt * 16 + nt) * 512 + lane * 16);
      v16bf bD = load_b(sWd2 + (kt * 16 + nt) * 512 + lane * 16);
      as = WMMA_BF16(load_a(aS, HD, kt, lane), bS, as);
      ap = WMMA_BF16(load_a(aP, HD, kt, lane), bD, ap);
      an = WMMA_BF16(load_a(aN, HD, kt, lane), bD, an);
    }
    float bsv = bs[nt * 16 + col], bdv = bdst[nt * 16 + col], wo = Wo[nt * 16 + col];
    #pragma unroll
    for (int v = 0; v < 8; ++v) {
      float hs = as[v] + bsv;
      pacc[v] += fmaxf(hs + ap[v] + bdv, 0.f) * wo;
      nacc[v] += fmaxf(hs + an[v] + bdv, 0.f) * wo;
    }
  }
  #pragma unroll
  for (int v = 0; v < 8; ++v) {
    #pragma unroll
    for (int m = 1; m < 16; m <<= 1) {
      pacc[v] += __shfl_xor(pacc[v], m, 32);
      nacc[v] += __shfl_xor(nacc[v], m, 32);
    }
  }
  if (col == 0) {
    #pragma unroll
    for (int v = 0; v < 8; ++v) {
      sRed[0][wave * 16 + half * 8 + v] = pacc[v];
      sRed[1][wave * 16 + half * 8 + v] = nacc[v];
    }
  }
  __syncthreads();
  if (tid < 2) {
    float sum = 0.f;
    for (int i = 0; i < 64; ++i) sum += sRed[tid][i];
    out[tid * 64 + blockIdx.x] = sum / 64.f + bo[0];
  }
}

// ---------------------------------------------------------------------------
extern "C" void kernel_launch(void* const* d_in, const int* in_sizes, int n_in,
                              void* d_out, int out_size, void* d_ws, size_t ws_size,
                              hipStream_t stream) {
  const float* x    = (const float*)d_in[0];
  const float* z    = (const float*)d_in[1];
  const float* ts   = (const float*)d_in[2];
  const float* w_t  = (const float*)d_in[3];
  const float* b_t  = (const float*)d_in[4];
  const float* W1   = (const float*)d_in[5];
  const float* b1   = (const float*)d_in[6];
  const float* W2   = (const float*)d_in[7];
  const float* b2   = (const float*)d_in[8];
  const float* W3   = (const float*)d_in[9];
  const float* b3   = (const float*)d_in[10];
  const float* Wd   = (const float*)d_in[11];
  const float* bd   = (const float*)d_in[12];
  const float* Wsm  = (const float*)d_in[13];
  const float* bs   = (const float*)d_in[14];
  const float* Wdst = (const float*)d_in[15];
  const float* bdst = (const float*)d_in[16];
  const float* Wo   = (const float*)d_in[17];
  const float* bo   = (const float*)d_in[18];

  char* ws = (char*)d_ws;
  __bf16* gW1p   = (__bf16*)(ws + 0);         //  64 KB
  __bf16* gW2p   = (__bf16*)(ws + 65536);     // 128 KB
  __bf16* gW3p   = (__bf16*)(ws + 196608);    //  64 KB
  __bf16* gWdp   = (__bf16*)(ws + 262144);    // 160 KB
  __bf16* gWsp   = (__bf16*)(ws + 425984);    // 128 KB
  __bf16* gWdstp = (__bf16*)(ws + 557056);    // 128 KB
  __bf16* gFeat  = (__bf16*)(ws + 688128);    // 12288*320 bf16 = 7.5 MB
  __bf16* gH     = (__bf16*)(ws + 8552448);   // 12288*256 bf16 = 6 MB

  // Pack weights into WMMA B-fragment layouts (bf16).
  pack_b_kernel<<<(64 * 32 + 255) / 256, 256, 0, stream>>>(W1, gW1p, 128, 128, 256);
  pack_b_kernel<<<(128 * 32 + 255) / 256, 256, 0, stream>>>(W2, gW2p, 256, 256, 256);
  pack_b_kernel<<<(64 * 32 + 255) / 256, 256, 0, stream>>>(W3, gW3p, 256, 256, 128);
  pack_b_kernel<<<(160 * 32 + 255) / 256, 256, 0, stream>>>(Wd, gWdp, 300, 320, 256);
  pack_b_kernel<<<(128 * 32 + 255) / 256, 256, 0, stream>>>(Wsm, gWsp, 256, 256, 256);
  pack_b_kernel<<<(128 * 32 + 255) / 256, 256, 0, stream>>>(Wdst, gWdstp, 256, 256, 256);

  featx_kernel<<<(NROWS * 192 + 255) / 256, 256, 0, stream>>>(x, gFeat);

  ode_kernel<<<NROWS / 64, 128, 0, stream>>>(z, ts, w_t, b_t, b1, b2, b3,
                                             gW1p, gW2p, gW3p, gFeat);
  decode_kernel<<<NROWS / 64, 128, 0, stream>>>(gFeat, gWdp, bd, gH);
  edge_kernel<<<64, 128, 0, stream>>>(gH, gWsp, gWdstp, bs, bdst, Wo, bo, (float*)d_out);
}
